// WeightedNormalizedBoundaryLoss_34497177321488
// MI455X (gfx1250) — matrix operand order repaired and checked by reading, not compiled, not run on hardware
//
#include <hip/hip_runtime.h>

// ---------------------------------------------------------------------------
// WeightedNormalizedBoundaryLoss for MI455X (gfx1250), wave32 + WMMA.
//
// distance_map = sqrt(Sv^2 + Sh^2) where Sv/Sh are inclusive suffix sums of
// the binary mask t along H / W (the cumsum/flip chain telescopes to this).
// den per plane = sum(dm) since ((1-t)^2 - t)^2 == 1 for binary t.
//
// P1a: segmented column suffix scan (exact in f16, values <= 128 per segment)
// P1b: in-place suffix-fix of segment sums -> per-(plane,seg,col) offsets,
//      plus exact global count of ones (class weights).
// P2 : per 16-row block, walk 16-col tiles right->left. Everything on the
//      XDL pipe, three v_wmma_f32_16x16x32_f16 per tile, no LDS shuffles:
//        d     = X * Ts + carry   (Ts[k][j] = k>=j)  -> Sh[i][j]
//        d2    = X * Tx + carry   (Tx[k][j] = k> j)  -> Sh[i][j+1], so t=d-d2
//        carry = X * 1  + carry   (rowsum broadcast == next carry C-matrix)
//      All integer-valued, exact in f16xf16->f32. Loss terms fused on VALU.
// P3 : tiny deterministic single-wave combine.
// ---------------------------------------------------------------------------

typedef __attribute__((ext_vector_type(16))) _Float16 v16h;
typedef __attribute__((ext_vector_type(8)))  float    v8f;

#define HW 1024
#define TOTAL_ELEMS 33554432.0f

// ---------------- Pass 1a: segmented column suffix scan ----------------
// grid: 32 planes * 8 segs * 4 colblocks = 1024 blocks, 256 threads
__global__ __launch_bounds__(256) void p1_colscan(
    const int* __restrict__ target, _Float16* __restrict__ sv16,
    float* __restrict__ segsum)
{
  const int b   = blockIdx.x;
  const int p   = b >> 5;
  const int rem = b & 31;
  const int seg = rem >> 2;
  const int j   = ((rem & 3) << 8) + threadIdx.x;
  const size_t base = ((size_t)p << 20) + (size_t)j;
  const int i0 = seg << 7;

  float acc = 0.0f;
#pragma unroll 8
  for (int i = i0 + 127; i >= i0; --i) {
    acc += (float)target[base + (size_t)i * HW];       // coalesced row reads
    sv16[base + (size_t)i * HW] = (_Float16)acc;       // exact: acc <= 128
  }
  segsum[((p << 3) + seg) * HW + j] = acc;
}

// ---------------- Pass 1b: suffix of segment sums (in place) + counts ------
// grid: 128 blocks * 256 threads = 32768 = planes * W columns
__global__ __launch_bounds__(256) void p1_fix(
    float* __restrict__ segsum, float* __restrict__ countpart)
{
  const int t = blockIdx.x * 256 + threadIdx.x;
  const int p = t >> 10;
  const int j = t & 1023;
  float* bp = segsum + ((size_t)(p << 3)) * HW + j;

  float s[8];
#pragma unroll
  for (int k = 0; k < 8; ++k) s[k] = bp[k * HW];
  float run = 0.0f;                       // exclusive suffix of segment sums
#pragma unroll
  for (int k = 7; k >= 0; --k) { float v = s[k]; bp[k * HW] = run; run += v; }
  // run == full column sum (count of ones in this column)

  __shared__ float red[256];
  red[threadIdx.x] = run;
  __syncthreads();
  for (int off = 128; off > 0; off >>= 1) {
    if (threadIdx.x < off) red[threadIdx.x] += red[threadIdx.x + off];
    __syncthreads();
  }
  if (threadIdx.x == 0) countpart[blockIdx.x] = red[0];
}

// ---------------- Pass 2: WMMA row suffix scan + fused loss ----------------
// grid: 32 planes * 64 rowblocks = 2048 blocks, 32 threads (one wave)
__global__ __launch_bounds__(32) void p2_rowscan_wmma(
    const float* __restrict__ pred, const int* __restrict__ target,
    const _Float16* __restrict__ sv16, const float* __restrict__ segoff,
    float* __restrict__ numpart, float* __restrict__ denpart)
{
  const int p    = blockIdx.x >> 6;
  const int rb   = blockIdx.x & 63;
  const int R0   = rb << 4;
  const int seg  = R0 >> 7;
  const int lane = threadIdx.x;
  const int half = lane >> 4;            // K half for A / row half for C,D
  const int N    = lane & 15;            // column (B,C,D) / row (A)
  const size_t pbase = (size_t)p << 20;

  // Constant B operands, zero-padded to K=32.
  // 16-bit B 32x16 layout: lane half h, elem e<8 <-> K=8h+e (K=0..15),
  // elems 8..15 hold K=16..31 (all zero here).
  const _Float16 one = (_Float16)1.0f, zero = (_Float16)0.0f;
  v16h b_incl, b_excl, b_ones;           // k>=j, k>j, all-ones (k<16)
#pragma unroll
  for (int e = 0; e < 16; ++e) {
    b_incl[e] = (e < 8 && (half << 3) + e >= N) ? one : zero;
    b_excl[e] = (e < 8 && (half << 3) + e >  N) ? one : zero;
    b_ones[e] = (e < 8)               ? one : zero;
  }

  v8f carry = {};                        // C-matrix: carry[i] uniform in j
  float num = 0.0f, den = 0.0f;
  const int arow = R0 + N;               // A-matrix row M
  const float* offrow = segoff + (size_t)((p << 3) + seg) * HW;
  const int* arow_ptr = target + pbase + (size_t)arow * HW;

#pragma unroll 2
  for (int tb = 63; tb >= 0; --tb) {
    const int colbase = tb << 4;

    // A tile: lane holds X[arow][colbase + 8*half + 0..7] in elems 0..7,
    // elems 8..15 (K>=16) zero. Two 16B loads + int->f16 convert.
    const int* tp = arow_ptr + colbase + (half << 3);
    if (tb) __builtin_prefetch((const char*)tp - 64, 0, 1);  // next tile
    int4 q0 = ((const int4*)tp)[0];
    int4 q1 = ((const int4*)tp)[1];
    v16h amat = {};
    amat[0] = (_Float16)q0.x; amat[1] = (_Float16)q0.y;
    amat[2] = (_Float16)q0.z; amat[3] = (_Float16)q0.w;
    amat[4] = (_Float16)q1.x; amat[5] = (_Float16)q1.y;
    amat[6] = (_Float16)q1.z; amat[7] = (_Float16)q1.w;

    // Sh[i][j] (inclusive suffix + carry)
    v8f d = __builtin_amdgcn_wmma_f32_16x16x32_f16(
        false, amat, false, b_incl, (short)0, carry, false, false);
    // Sh[i][j+1] (strict suffix + carry; j==15 column is pure carry)
    v8f d2 = __builtin_amdgcn_wmma_f32_16x16x32_f16(
        false, amat, false, b_excl, (short)0, carry, false, false);
    // carry[i] += rowsum(X)[i], broadcast across j: next tile's C-matrix
    carry = __builtin_amdgcn_wmma_f32_16x16x32_f16(
        false, amat, false, b_ones, (short)0, carry, false, false);

    const float offj = offrow[colbase + N];   // column-scan segment offset

#pragma unroll
    for (int r = 0; r < 8; ++r) {
      const float sh   = d[r];                 // Sh at (row, colbase+N)
      const float tval = sh - d2[r];           // t, exact, no shuffles
      const int row = R0 + (half << 3) + r;    // C/D row mapping
      const size_t idx = pbase + (size_t)row * HW + colbase + N;
      const float sv = (float)sv16[idx] + offj;
      const float dm = sqrtf(sh * sh + sv * sv);
      const float diff = (1.0f - tval) - pred[idx];  // (1-t)^2 == 1-t
      num = fmaf(dm * diff, diff, num);
      den += dm;
    }
  }

  // wave32 reduction, lane 0 owns this (plane,rowblock) partial (no atomics)
  for (int off = 16; off > 0; off >>= 1) {
    num += __shfl_down(num, off, 32);
    den += __shfl_down(den, off, 32);
  }
  if (lane == 0) { numpart[blockIdx.x] = num; denpart[blockIdx.x] = den; }
}

// ---------------- Pass 3: deterministic single-wave combine ----------------
__global__ __launch_bounds__(32) void p3_final(
    const float* __restrict__ numpart, const float* __restrict__ denpart,
    const float* __restrict__ countpart, float* __restrict__ out)
{
  const int lane = threadIdx.x;
  float c1 = 0.0f;
  for (int k = 0; k < 128; ++k) c1 += countpart[k];    // count of ones
  const float c0  = TOTAL_ELEMS - c1;
  const float cw0 = (TOTAL_ELEMS / c0) * (TOTAL_ELEMS / c0);
  const float cw1 = (TOTAL_ELEMS / c1) * (TOTAL_ELEMS / c1);

  float contrib = 0.0f;
  if (lane < 16) {                                     // lane = batch index b
    float numb = 0.0f, denb = 0.0f;
#pragma unroll
    for (int c = 0; c < 2; ++c) {
      const float cw = (c == 0) ? cw0 : cw1;
      float sn = 0.0f, sd = 0.0f;
      const int base = (lane * 2 + c) * 64;
      for (int k = 0; k < 64; ++k) { sn += numpart[base + k]; sd += denpart[base + k]; }
      numb += cw * sn; denb += cw * sd;
    }
    contrib = 1.0f - numb / denb;
  }
  for (int off = 16; off > 0; off >>= 1) contrib += __shfl_down(contrib, off, 32);
  if (lane == 0) out[0] = contrib / 16.0f;             // mean over B
}

// ---------------------------------------------------------------------------
extern "C" void kernel_launch(void* const* d_in, const int* in_sizes, int n_in,
                              void* d_out, int out_size, void* d_ws, size_t ws_size,
                              hipStream_t stream) {
  const float* pred   = (const float*)d_in[0];
  const int*   target = (const int*)d_in[1];

  float* ws        = (float*)d_ws;
  float* numpart   = ws;                // 2048 f32
  float* denpart   = ws + 2048;         // 2048 f32
  float* countpart = ws + 4096;         //  128 f32
  float* segsum    = ws + 8192;         // 32*8*1024 f32 = 1 MiB (reused as offsets)
  _Float16* sv16   = (_Float16*)((char*)d_ws + (4ull << 20));  // 64 MiB

  p1_colscan<<<1024, 256, 0, stream>>>(target, sv16, segsum);
  p1_fix<<<128, 256, 0, stream>>>(segsum, countpart);
  p2_rowscan_wmma<<<2048, 32, 0, stream>>>(pred, target, sv16, segsum,
                                           numpart, denpart);
  p3_final<<<1, 32, 0, stream>>>(numpart, denpart, countpart, (float*)d_out);
}
